// knn_contrastive_loss_31628139168315
// MI455X (gfx1250) — matrix operand order repaired and checked by reading, not compiled, not run on hardware
//
#include <hip/hip_runtime.h>

#define NN    16384
#define DD    24
#define KK    10
#define DPAD  32
#define TS    20          // padded LDS row stride (words): 16B-aligned rows, conflict-free

typedef __attribute__((ext_vector_type(16))) _Float16 v16h;
typedef __attribute__((ext_vector_type(8)))  _Float16 v8h;
typedef __attribute__((ext_vector_type(8)))  float    v8f;
typedef __attribute__((ext_vector_type(4)))  float    v4f;

// ---------------- prep: f32[N][24] -> f16[N][32] (zero-padded) + row norms ----
__global__ __launch_bounds__(256) void prep_kernel(const float* __restrict__ E,
                                                   _Float16* __restrict__ Eh,
                                                   float* __restrict__ sq) {
  int row = blockIdx.x * blockDim.x + threadIdx.x;
  if (row >= NN) return;
  const float* src = E + (size_t)row * DD;
  _Float16* dst = Eh + (size_t)row * DPAD;
  float s = 0.f;
#pragma unroll
  for (int d = 0; d < DD; ++d) {
    float x = src[d];
    _Float16 h = (_Float16)x;
    float xr = (float)h;        // norm of the rounded value -> consistent with WMMA dot
    s += xr * xr;
    dst[d] = h;
  }
#pragma unroll
  for (int d = DD; d < DPAD; ++d) dst[d] = (_Float16)0.f;
  sq[row] = s;
}

// ------------- fused WMMA distance tiles + streaming top-10 + hinge loss ------
// One wave (32 threads) per 16-row strip; 2 column tiles (32 cols) per iter.
// Lane group g scans tile g -> every lane does selection; lists merged at end.
__global__ __launch_bounds__(32) void knn_loss_kernel(
    const _Float16* __restrict__ Eh, const float* __restrict__ sq,
    const int* __restrict__ pid, const float* __restrict__ marginp,
    float* __restrict__ partial) {
  __shared__ float tile[2][16 * TS];
  __shared__ float mergeD[16 * KK];
  __shared__ int   mergeI[16 * KK];

  const int lane = threadIdx.x;             // 0..31 (wave32)
  const int g    = lane >> 4;               // half-wave selector
  const int ln   = lane & 15;
  const int rowbase = blockIdx.x * 16;

  // A fragment: 16x32 f16. Lane group g holds K in {16g..16g+15}.
  const _Float16* arow = Eh + (size_t)(rowbase + ln) * DPAD;
  v8h alo = *(const v8h*)(arow + 16 * g);
  v8h ahi = *(const v8h*)(arow + 16 * g + 8);
  v16h a = __builtin_shufflevector(alo, ahi,
                                   0, 1, 2, 3, 4, 5, 6, 7,
                                   8, 9, 10, 11, 12, 13, 14, 15);

  // Row norms matching the C-tile layout: C vgpr v of lane holds row v + 8g.
  float sqr[8];
#pragma unroll
  for (int v = 0; v < 8; ++v) sqr[v] = sq[rowbase + v + 8 * g];

  float bestD[KK];
  int   bestI[KK];
#pragma unroll
  for (int t = 0; t < KK; ++t) { bestD[t] = 3.0e38f; bestI[t] = 0; }

  for (int cb = 0; cb < NN; cb += 32) {
    // B fragments for two column tiles: B = E^T, 32 contiguous bytes per lane.
    const _Float16* brow0 = Eh + (size_t)(cb + ln) * DPAD;
    const _Float16* brow1 = Eh + (size_t)(cb + 16 + ln) * DPAD;
    v16h b0 = *(const v16h*)(brow0 + 16 * g);
    v16h b1 = *(const v16h*)(brow1 + 16 * g);
    if (cb + 32 < NN) {   // near-scope prefetch of next 32-row B panel (L2-resident)
      __builtin_prefetch((const void*)(Eh + (size_t)(cb + 32 + ln) * DPAD), 0, 3);
      __builtin_prefetch((const void*)(Eh + (size_t)(cb + 48 + ln) * DPAD), 0, 3);
    }

    v8f c0 = {}, c1 = {};
    c0 = __builtin_amdgcn_wmma_f32_16x16x32_f16(false, a, false, b0,
                                                (short)0, c0, false, false);
    c1 = __builtin_amdgcn_wmma_f32_16x16x32_f16(false, a, false, b1,
                                                (short)0, c1, false, false);

    float sqc0 = sq[cb + ln];
    float sqc1 = sq[cb + 16 + ln];
    float dd0[8], dd1[8];
#pragma unroll
    for (int v = 0; v < 8; ++v) {
      dd0[v] = __builtin_fmaf(-2.0f, c0[v], sqr[v] + sqc0);
      dd1[v] = __builtin_fmaf(-2.0f, c1[v], sqr[v] + sqc1);
    }

    // self-loop exclusion: diagonal hits exactly one of the two tiles, once.
    if (cb == rowbase) {
#pragma unroll
      for (int v = 0; v < 8; ++v)
        if (v + 8 * g == ln) dd0[v] = 3.0e38f;
    }
    if (cb + 16 == rowbase) {
#pragma unroll
      for (int v = 0; v < 8; ++v)
        if (v + 8 * g == ln) dd1[v] = 3.0e38f;
    }

#pragma unroll
    for (int v = 0; v < 8; ++v) {
      tile[0][(v + 8 * g) * TS + ln] = dd0[v];
      tile[1][(v + 8 * g) * TS + ln] = dd1[v];
    }
    __syncthreads();            // single-wave WG: just the ds ordering fence

    // Lane group g scans tile g: row ln, 16 candidates via 4x ds_load_b128.
    {
      const v4f* rp = (const v4f*)(&tile[g][ln * TS]);
      const int colbase = cb + 16 * g;
#pragma unroll
      for (int j = 0; j < 4; ++j) {
        v4f q = rp[j];
#pragma unroll
        for (int e = 0; e < 4; ++e) {
          float val = q[e];
          if (val < bestD[KK - 1]) {        // threshold gate: 1 cmp steady-state
            bestD[KK - 1] = val;
            bestI[KK - 1] = colbase + 4 * j + e;
#pragma unroll
            for (int s2 = KK - 1; s2 > 0; --s2) {
              if (bestD[s2] < bestD[s2 - 1]) {
                float td = bestD[s2]; bestD[s2] = bestD[s2 - 1]; bestD[s2 - 1] = td;
                int   ti = bestI[s2]; bestI[s2] = bestI[s2 - 1]; bestI[s2 - 1] = ti;
              }
            }
          }
        }
      }
    }
    __syncthreads();
  }

  // ---- one-time merge: group-1 list for row ln joins group-0 list ----
  if (g == 1) {
#pragma unroll
    for (int t = 0; t < KK; ++t) {
      mergeD[ln * KK + t] = bestD[t];
      mergeI[ln * KK + t] = bestI[t];
    }
  }
  __syncthreads();
  float sum = 0.f;
  if (g == 0) {
#pragma unroll
    for (int t = 0; t < KK; ++t) {
      float val = mergeD[ln * KK + t];
      int   idx = mergeI[ln * KK + t];
      if (val < bestD[KK - 1]) {
        bestD[KK - 1] = val;
        bestI[KK - 1] = idx;
#pragma unroll
        for (int s2 = KK - 1; s2 > 0; --s2) {
          if (bestD[s2] < bestD[s2 - 1]) {
            float td = bestD[s2]; bestD[s2] = bestD[s2 - 1]; bestD[s2 - 1] = td;
            int   ti = bestI[s2]; bestI[s2] = bestI[s2 - 1]; bestI[s2 - 1] = ti;
          }
        }
      }
    }
    // hinge contrastive loss over this row's K edges
    float m  = marginp[0];
    float m2 = m * m;
    int myPid = pid[rowbase + ln];
#pragma unroll
    for (int t = 0; t < KK; ++t) {
      float d = bestD[t];
      float hinge = fmaxf(m2 - d, 0.f);
      sum += (pid[bestI[t]] == myPid) ? d : hinge;
    }
  }
  // wave32 reduction (lanes 16..31 contribute 0)
#pragma unroll
  for (int off = 16; off > 0; off >>= 1) sum += __shfl_xor(sum, off, 32);
  if (lane == 0) partial[blockIdx.x] = sum;
}

// ---------------- deterministic final reduction: 1024 partials -> mean --------
__global__ __launch_bounds__(256) void reduce_kernel(const float* __restrict__ partial,
                                                     float* __restrict__ out) {
  __shared__ float s[256];
  int t = threadIdx.x;
  float v = partial[t] + partial[t + 256] + partial[t + 512] + partial[t + 768];
  s[t] = v;
  __syncthreads();
  for (int o = 128; o > 0; o >>= 1) {
    if (t < o) s[t] += s[t + o];
    __syncthreads();
  }
  if (t == 0) out[0] = s[0] * (1.0f / ((float)NN * (float)KK));
}

extern "C" void kernel_launch(void* const* d_in, const int* in_sizes, int n_in,
                              void* d_out, int out_size, void* d_ws, size_t ws_size,
                              hipStream_t stream) {
  (void)in_sizes; (void)n_in; (void)out_size; (void)ws_size;
  const float* E      = (const float*)d_in[0];
  const int*   pid    = (const int*)d_in[1];
  const float* margin = (const float*)d_in[2];
  // d_in[3] = k (always 10; compile-time constant here)

  char* ws = (char*)d_ws;
  _Float16* Eh      = (_Float16*)ws;                                  // 1 MiB
  float*    sq      = (float*)(ws + (size_t)NN * DPAD * 2);           // 64 KiB
  float*    partial = (float*)(ws + (size_t)NN * DPAD * 2 + (size_t)NN * 4); // 4 KiB

  prep_kernel<<<NN / 256, 256, 0, stream>>>(E, Eh, sq);
  knn_loss_kernel<<<NN / 16, 32, 0, stream>>>(Eh, sq, pid, margin, partial);
  reduce_kernel<<<1, 256, 0, stream>>>(partial, (float*)d_out);
}
